// Decoupled_Block_20779051778112
// MI455X (gfx1250) — matrix operand order repaired and checked by reading, not compiled, not run on hardware
//
#include <hip/hip_runtime.h>
#include <hip/hip_bf16.h>

// ---------------------------------------------------------------------------
// Problem constants (reference: B=2, S=1024, D=1024, NS=2, H=16)
// ---------------------------------------------------------------------------
constexpr int B_  = 2;
constexpr int S_  = 1024;
constexpr int D_  = 1024;
constexpr int NS_ = 2;
constexpr int DH_ = 64;                 // D / H
constexpr int BS_ = B_ * S_;            // 2048 rows
constexpr long BSD_ = (long)BS_ * D_;   // elements per activation tensor
constexpr long DD_  = (long)D_ * D_;    // elements per weight matrix

typedef _Float16 half_t;
typedef __attribute__((ext_vector_type(8)))  _Float16 h8;
typedef __attribute__((ext_vector_type(16))) _Float16 v16h;
typedef __attribute__((ext_vector_type(8)))  float    v8f;

// V_WMMA_F32_16X16X32_F16 : D(16x16,f32) = A(16x32,f16) * B(32x16,f16) + C
#define WMMA16(a, b, c) \
  __builtin_amdgcn_wmma_f32_16x16x32_f16(false, (a), false, (b), (short)0, (c), false, false)

__device__ __forceinline__ v16h cat8(h8 lo, h8 hi) {
  return __builtin_shufflevector(lo, hi, 0,1,2,3,4,5,6,7,8,9,10,11,12,13,14,15);
}

__device__ __forceinline__ float gelu_erf(float x) {
  return 0.5f * x * (1.0f + erff(x * 0.70710678118654752f));
}

// Async DMA: global -> LDS, 16B per lane, tracked by ASYNCcnt (no VGPR data).
// Generic pointers to LDS carry the LDS byte offset in their low 32 bits.
__device__ __forceinline__ void async_copy_b128(void* lds_ptr, const void* gptr) {
  const unsigned lds_off = (unsigned)(unsigned long long)lds_ptr;
  asm volatile("global_load_async_to_lds_b128 %0, %1, off"
               :: "v"(lds_off), "v"(gptr) : "memory");
}
__device__ __forceinline__ void wait_asynccnt0() {
  asm volatile("s_wait_asynccnt 0" ::: "memory");
}

// ---------------------------------------------------------------------------
// fp32 -> f16 conversion (weights / inputs), grid-stride.
// ---------------------------------------------------------------------------
__global__ __launch_bounds__(256)
void f32_to_f16_kernel(const float* __restrict__ src, half_t* __restrict__ dst, long n) {
  long i = (long)blockIdx.x * blockDim.x + threadIdx.x;
  const long stride = (long)gridDim.x * blockDim.x;
  for (; i < n; i += stride) dst[i] = (half_t)src[i];
}

// ---------------------------------------------------------------------------
// f16 GEMM, fp32 accumulate:
//   acc[M,N] = A16[M,K] @ W16[K,N];  out = act(acc + bias (+ resid))
//   Block 256 threads (8 waves), tile 64x128, K chunk 64, DOUBLE-BUFFERED:
//     - A tile staged by GLOBAL_LOAD_ASYNC_TO_LDS_B128 (ASYNCcnt DMA path)
//     - W tile loaded to VGPRs early, scatter-TRANSPOSED to LDS after compute
//   Wave w: subtile-row (w&3), 4 subtile-cols (w>>2)*4 .. +3 (A-frag reuse x4).
// ---------------------------------------------------------------------------
__global__ __launch_bounds__(256)
void gemm16_kernel(const half_t* __restrict__ A, long lda,
                   const half_t* __restrict__ W,
                   const float* __restrict__ bias,
                   const float* __restrict__ resid,
                   float* __restrict__ out32,
                   half_t* __restrict__ out16,
                   int M, int N, int K, int act) {
  constexpr int LDA_S = 72;   // f16 elems per As row   (64 + pad)
  constexpr int LDB_S = 72;   // f16 elems per Bst row
  __shared__ half_t As[2][64 * LDA_S];     // As[m][k]
  __shared__ half_t Bst[2][128 * LDB_S];   // Bst[n][k] (transposed W tile)

  const int tid  = threadIdx.x;
  const int lane = tid & 31;
  const int wave = tid >> 5;
  const int l16  = lane & 15;
  const int lh   = (lane < 16) ? 0 : 1;     // lane half

  const int bm = blockIdx.y * 64;
  const int bn = blockIdx.x * 128;
  const int sm = wave & 3;                  // subtile row (0..3)
  const int cg = wave >> 2;                 // col group (0/1): cols cg*64..+63

  // ---- staging helpers ----
  auto issueA = [&](int k0, int buf) {      // 64x64 f16 tile via async DMA
#pragma unroll
    for (int j = 0; j < 2; ++j) {
      const int f  = tid + j * 256;         // 0..511
      const int r  = f >> 3;
      const int c8 = (f & 7) << 3;
      async_copy_b128(&As[buf][r * LDA_S + c8],
                      A + (long)(bm + r) * lda + (k0 + c8));
    }
  };
  auto loadB = [&](int k0, h8* breg) {      // 64x128 f16 tile -> registers
#pragma unroll
    for (int j = 0; j < 4; ++j) {
      const int f  = tid + j * 256;         // 0..1023
      const int r  = f >> 4;                // k row (64)
      const int c8 = (f & 15) << 3;         // n chunk (128/8)
      breg[j] = *(const h8*)(W + (long)(k0 + r) * N + (bn + c8));
    }
  };
  auto storeB = [&](const h8* breg, int buf) {   // scatter transpose to LDS
#pragma unroll
    for (int j = 0; j < 4; ++j) {
      const int f  = tid + j * 256;
      const int r  = f >> 4;
      const int c8 = (f & 15) << 3;
#pragma unroll
      for (int i = 0; i < 8; ++i) Bst[buf][(c8 + i) * LDB_S + r] = breg[j][i];
    }
  };

  v8f acc[4] = { v8f{}, v8f{}, v8f{}, v8f{} };
  h8 breg[4];

  // ---- prologue: stage chunk 0 ----
  issueA(0, 0);
  loadB(0, breg);
  storeB(breg, 0);
  wait_asynccnt0();
  __syncthreads();

  for (int k0 = 0; k0 < K; k0 += 64) {
    const int cur  = (k0 >> 6) & 1;
    const bool more = (k0 + 64) < K;

    // issue next chunk's loads BEFORE compute (overlap with WMMA)
    if (more) { issueA(k0 + 64, cur ^ 1); loadB(k0 + 64, breg); }

    // ---- 2 WMMA K-steps (K=32 each); A-frag reused across 4 cols ----
    const half_t* as = &As[cur][0];
    const half_t* bs = &Bst[cur][0];
#pragma unroll
    for (int ks = 0; ks < 2; ++ks) {
      const int arow = sm * 16 + l16;
      const int c0   = ks * 32 + lh * 16;
      const v16h af = cat8(*(const h8*)&as[arow * LDA_S + c0],
                           *(const h8*)&as[arow * LDA_S + c0 + 8]);
      const int kb = ks * 32 + lh * 16;
#pragma unroll
      for (int t = 0; t < 4; ++t) {
        const int col = (cg * 4 + t) * 16 + l16;
        const v16h bf = cat8(*(const h8*)&bs[col * LDB_S + kb],
                             *(const h8*)&bs[col * LDB_S + kb + 8]);
        acc[t] = WMMA16(af, bf, acc[t]);
      }
    }

    // transpose-scatter next B tile after compute (loads already in flight)
    if (more) storeB(breg, cur ^ 1);

    wait_asynccnt0();        // next A tile resident (compiler can't see DMA)
    __syncthreads();
  }

  // ---- epilogue: bias (+resid) (+gelu); write f32 and/or f16 ----
#pragma unroll
  for (int t = 0; t < 4; ++t) {
    const int colg = bn + (cg * 4 + t) * 16 + l16;
    const float bv = bias[colg];
#pragma unroll
    for (int r = 0; r < 8; ++r) {
      const int rowl = (lane < 16) ? r : (r + 8);
      const long rowg = bm + sm * 16 + rowl;
      float vo = acc[t][r] + bv;
      if (resid) vo += resid[rowg * N + colg];
      if (act == 1) vo = gelu_erf(vo);
      if (out32) out32[rowg * N + colg] = vo;
      if (out16) out16[rowg * N + colg] = (half_t)vo;
    }
  }
}

// ---------------------------------------------------------------------------
// Fused 3-branch attention (flash-style), f16 operands / fp32 accumulate.
// One wave per (b, h, 16-row q-tile).
//   Pass 1: per-branch online (rowmax,rowsum), scores via WMMA f16 (2/tile).
//   Pass 2: k-tiles in PAIRS: recompute scores, p=exp(s-m)/sum per branch,
//           A = max over branches -> f16 16x32 tile in LDS -> O += A @ V.
// Never materializes the (B,H,S,S) maps (saves ~0.8 GB of HBM traffic).
// ---------------------------------------------------------------------------
__global__ void attn_fused_kernel(const half_t* __restrict__ q,
                                  const half_t* __restrict__ k,
                                  const half_t* __restrict__ v,
                                  const half_t* __restrict__ sq,
                                  const half_t* __restrict__ sk,
                                  half_t* __restrict__ o) {
  constexpr int TLD = 48;                      // f16 stride of A-tile buffer
  __shared__ half_t tposed[8][16 * TLD];       // per-wave 16x32 A-tile (f16)

  const int tid  = threadIdx.x;
  const int lane = tid & 31;
  const int wave = tid >> 5;
  const int l16  = lane & 15;
  const int lh   = (lane < 16) ? 0 : 1;
  half_t* tbuf = tposed[wave];

  const int wid = blockIdx.x * 8 + wave;       // 0 .. 2047
  const int qt  = wid & 63;
  const int h   = (wid >> 6) & 15;
  const int b   = wid >> 10;

  const float scale = 0.03125f;                // 1/sqrt(D) = 1/32

  const half_t* qb[3] = { q, sq, sq + BSD_ };
  const half_t* kb[3] = { k, sk, sk + BSD_ };

  const long qrow0 = (long)b * S_ + qt * 16;

  // ---- preload Q A-fragments: 3 branches x 2 K-chunks (dh 0-31, 32-63) ----
  v16h qa[3][2];
#pragma unroll
  for (int br = 0; br < 3; ++br) {
    const half_t* Q = qb[br] + (qrow0 + l16) * D_ + h * DH_;
#pragma unroll
    for (int kc = 0; kc < 2; ++kc) {
      const int c0 = kc * 32 + lh * 16;
      qa[br][kc] = cat8(*(const h8*)(Q + c0), *(const h8*)(Q + c0 + 8));
    }
  }

  // ---- pass 1: per-branch online softmax statistics ----
  float rm[3][8], rs[3][8];
#pragma unroll
  for (int br = 0; br < 3; ++br)
#pragma unroll
    for (int r = 0; r < 8; ++r) { rm[br][r] = -1e30f; rs[br][r] = 0.f; }

  for (int kt = 0; kt < 64; ++kt) {
#pragma unroll
    for (int br = 0; br < 3; ++br) {
      // B = K^T fragment: contiguous 32B along dh from K row l16
      const half_t* K = kb[br] + ((long)b * S_ + kt * 16 + l16) * D_ + h * DH_;
      v8f s = {};
#pragma unroll
      for (int kc = 0; kc < 2; ++kc) {
        const int kbo = kc * 32 + lh * 16;
        const v16h bf = cat8(*(const h8*)(K + kbo), *(const h8*)(K + kbo + 8));
        s = WMMA16(qa[br][kc], bf, s);
      }
#pragma unroll
      for (int r = 0; r < 8; ++r) {
        const float sv = s[r] * scale;
        float tm = sv;                         // row reduce over 16-lane half
#pragma unroll
        for (int m = 1; m < 16; m <<= 1) tm = fmaxf(tm, __shfl_xor(tm, m, 32));
        const float mnew = fmaxf(rm[br][r], tm);
        float p = __expf(sv - mnew);
#pragma unroll
        for (int m = 1; m < 16; m <<= 1) p += __shfl_xor(p, m, 32);
        rs[br][r] = rs[br][r] * __expf(rm[br][r] - mnew) + p;
        rm[br][r] = mnew;
      }
    }
  }

  // ---- pass 2: max-of-softmax fusion + O accumulation, k-tiles in pairs ----
  v8f oacc[4] = { v8f{}, v8f{}, v8f{}, v8f{} };

  for (int p2 = 0; p2 < 32; ++p2) {
#pragma unroll
    for (int sub = 0; sub < 2; ++sub) {
      const int kt = p2 * 2 + sub;
      float am[8];
#pragma unroll
      for (int r = 0; r < 8; ++r) am[r] = 0.f;
#pragma unroll
      for (int br = 0; br < 3; ++br) {
        const half_t* K = kb[br] + ((long)b * S_ + kt * 16 + l16) * D_ + h * DH_;
        v8f s = {};
#pragma unroll
        for (int kc = 0; kc < 2; ++kc) {
          const int kbo = kc * 32 + lh * 16;
          const v16h bf = cat8(*(const h8*)(K + kbo), *(const h8*)(K + kbo + 8));
          s = WMMA16(qa[br][kc], bf, s);
        }
#pragma unroll
        for (int r = 0; r < 8; ++r) {
          const float pr = __expf(s[r] * scale - rm[br][r]) / rs[br][r];
          am[r] = fmaxf(am[r], pr);
        }
      }
      // C-layout -> row-major f16 [qrow][kcol 0..31] in per-wave LDS
#pragma unroll
      for (int r = 0; r < 8; ++r) {
        const int rowl = (lane < 16) ? r : (r + 8);
        tbuf[rowl * TLD + sub * 16 + l16] = (half_t)am[r];
      }
    }
    asm volatile("s_wait_dscnt 0" ::: "memory");    // same-wave LDS RAW fence

    // O(16x64) += A(16x32,f16) @ V(32x64,f16)  -- single K=32 WMMA step
    const v16h af = cat8(*(const h8*)&tbuf[l16 * TLD + lh * 16],
                         *(const h8*)&tbuf[l16 * TLD + lh * 16 + 8]);
    const half_t* V = v + ((long)b * S_ + p2 * 32) * D_ + h * DH_;
#pragma unroll
    for (int nt = 0; nt < 4; ++nt) {
      v16h bf;   // elem e <-> V row (e + 16*laneHalf), col nt*16 + l16
#pragma unroll
      for (int e = 0; e < 16; ++e)
        bf[e] = V[(long)(e + lh * 16) * D_ + nt * 16 + l16];
      oacc[nt] = WMMA16(af, bf, oacc[nt]);
    }
  }

  // ---- store O tile (f16, feeds the Wo GEMM) ----
  half_t* O = o + qrow0 * D_ + h * DH_;
#pragma unroll
  for (int nt = 0; nt < 4; ++nt)
#pragma unroll
    for (int r = 0; r < 8; ++r) {
      const int rowl = (lane < 16) ? r : (r + 8);
      O[(long)rowl * D_ + nt * 16 + l16] = (half_t)oacc[nt][r];
    }
}

// ---------------------------------------------------------------------------
// Row LayerNorm: one 256-thread block per row; writes f32 and optional f16.
// ---------------------------------------------------------------------------
__global__ __launch_bounds__(256)
void layernorm_kernel(const float* __restrict__ x, const float* __restrict__ g,
                      const float* __restrict__ be,
                      float* __restrict__ out32, half_t* __restrict__ out16, int N) {
  __shared__ float r1[256], r2[256];
  const long row = blockIdx.x;
  const float* xr = x + row * N;
  float s = 0.f, s2 = 0.f;
  for (int i = threadIdx.x; i < N; i += 256) {
    const float vv = xr[i];
    s += vv; s2 += vv * vv;
  }
  r1[threadIdx.x] = s; r2[threadIdx.x] = s2;
  __syncthreads();
  for (int off = 128; off > 0; off >>= 1) {
    if (threadIdx.x < off) {
      r1[threadIdx.x] += r1[threadIdx.x + off];
      r2[threadIdx.x] += r2[threadIdx.x + off];
    }
    __syncthreads();
  }
  const float mean = r1[0] * (1.0f / N);
  const float var  = r2[0] * (1.0f / N) - mean * mean;
  const float inv  = rsqrtf(var + 1e-5f);
  for (int i = threadIdx.x; i < N; i += 256) {
    const float vo = (xr[i] - mean) * inv * g[i] + be[i];
    if (out32) out32[row * (long)N + i] = vo;
    if (out16) out16[row * (long)N + i] = (half_t)vo;
  }
}

// ---------------------------------------------------------------------------
// Host-side pipeline
// ---------------------------------------------------------------------------
extern "C" void kernel_launch(void* const* d_in, const int* in_sizes, int n_in,
                              void* d_out, int out_size, void* d_ws, size_t ws_size,
                              hipStream_t stream) {
  (void)in_sizes; (void)n_in; (void)out_size; (void)ws_size;

  const float* id_x = (const float*)d_in[0];
  const float* side = (const float*)d_in[1];
  /* d_in[2] = num_heads (int scalar) — fixed at 16 */
  const float* Wq  = (const float*)d_in[3];
  const float* bq  = (const float*)d_in[4];
  const float* Wk  = (const float*)d_in[5];
  const float* bk  = (const float*)d_in[6];
  const float* Wv  = (const float*)d_in[7];
  const float* bv  = (const float*)d_in[8];
  const float* sWq = (const float*)d_in[9];
  const float* sbq = (const float*)d_in[10];
  const float* sWk = (const float*)d_in[11];
  const float* sbk = (const float*)d_in[12];
  const float* Wo  = (const float*)d_in[13];
  const float* bo  = (const float*)d_in[14];
  const float* W1  = (const float*)d_in[15];
  const float* b1  = (const float*)d_in[16];
  const float* W2  = (const float*)d_in[17];
  const float* b2  = (const float*)d_in[18];
  const float* g1  = (const float*)d_in[19];
  const float* be1 = (const float*)d_in[20];
  const float* g2  = (const float*)d_in[21];
  const float* be2 = (const float*)d_in[22];

  // ---- workspace carve-out ----
  char* p = (char*)d_ws;
  auto a16 = [&](long n) { half_t* r = (half_t*)p; p += n * sizeof(half_t); return r; };
  auto a32 = [&](long n) { float*  r = (float*)p;  p += n * sizeof(float);  return r; };

  half_t* idx16  = a16(BSD_);
  half_t* side16 = a16(2 * BSD_);
  half_t* wq16   = a16(DD_);
  half_t* wk16   = a16(DD_);
  half_t* wv16   = a16(DD_);
  half_t* swq16  = a16(2 * DD_);
  half_t* swk16  = a16(2 * DD_);
  half_t* wo16   = a16(DD_);
  half_t* w1_16  = a16(DD_);
  half_t* w2_16  = a16(DD_);
  half_t* q16    = a16(BSD_);
  half_t* k16    = a16(BSD_);
  half_t* v16    = a16(BSD_);
  half_t* sq16   = a16(2 * BSD_);
  half_t* sk16   = a16(2 * BSD_);
  half_t* o16    = a16(BSD_);
  half_t* xl16   = a16(BSD_);
  half_t* h16    = a16(BSD_);
  float*  x1B    = a32(BSD_);
  float*  xlB    = a32(BSD_);
  float*  x2B    = a32(BSD_);

  // ---- fp32 -> f16 conversions (weights + inputs) ----
  auto conv = [&](const float* s, half_t* d, long n) {
    f32_to_f16_kernel<<<512, 256, 0, stream>>>(s, d, n);
  };
  conv(id_x, idx16, BSD_);
  conv(side, side16, 2 * BSD_);
  conv(Wq, wq16, DD_);   conv(Wk, wk16, DD_);   conv(Wv, wv16, DD_);
  conv(sWq, swq16, 2 * DD_);  conv(sWk, swk16, 2 * DD_);
  conv(Wo, wo16, DD_);   conv(W1, w1_16, DD_);  conv(W2, w2_16, DD_);

  const dim3 gg(D_ / 128, BS_ / 64);   // (8, 32) blocks, tile 64x128

  // id-branch QKV projections (f16 outputs feed attention)
  gemm16_kernel<<<gg, 256, 0, stream>>>(idx16, D_, wq16, bq, nullptr, nullptr, q16, BS_, D_, D_, 0);
  gemm16_kernel<<<gg, 256, 0, stream>>>(idx16, D_, wk16, bk, nullptr, nullptr, k16, BS_, D_, D_, 0);
  gemm16_kernel<<<gg, 256, 0, stream>>>(idx16, D_, wv16, bv, nullptr, nullptr, v16, BS_, D_, D_, 0);

  // side-branch Q/K projections (strided rows through (B,S,NS,D))
  for (int n = 0; n < NS_; ++n) {
    gemm16_kernel<<<gg, 256, 0, stream>>>(side16 + n * D_, (long)NS_ * D_,
        swq16 + n * DD_, sbq + n * D_, nullptr, nullptr, sq16 + n * BSD_, BS_, D_, D_, 0);
    gemm16_kernel<<<gg, 256, 0, stream>>>(side16 + n * D_, (long)NS_ * D_,
        swk16 + n * DD_, sbk + n * D_, nullptr, nullptr, sk16 + n * BSD_, BS_, D_, D_, 0);
  }

  // fused 3-branch attention (B*H*S/16 = 2048 waves, 8 waves/block)
  attn_fused_kernel<<<256, 256, 0, stream>>>(q16, k16, v16, sq16, sk16, o16);

  // x = id_x + O @ Wo + bo ; LN1 (f32 + f16 copies)
  gemm16_kernel<<<gg, 256, 0, stream>>>(o16, D_, wo16, bo, id_x, x1B, nullptr, BS_, D_, D_, 0);
  layernorm_kernel<<<BS_, 256, 0, stream>>>(x1B, g1, be1, xlB, xl16, D_);

  // h = gelu(x @ W1 + b1) ; x = x + h @ W2 + b2 ; LN2 -> out
  gemm16_kernel<<<gg, 256, 0, stream>>>(xl16, D_, w1_16, b1, nullptr, nullptr, h16, BS_, D_, D_, 1);
  gemm16_kernel<<<gg, 256, 0, stream>>>(h16, D_, w2_16, b2, xlB, x2B, nullptr, BS_, D_, D_, 0);
  layernorm_kernel<<<BS_, 256, 0, stream>>>(x2B, g2, be2, (float*)d_out, nullptr, D_);
}
